// GATv2_57363583205417
// MI455X (gfx1250) — matrix operand order repaired and checked by reading, not compiled, not run on hardware
//
#include <hip/hip_runtime.h>
#include <hip/hip_bf16.h>

typedef __attribute__((ext_vector_type(16))) _Float16 v16h;
typedef __attribute__((ext_vector_type(8)))  _Float16 v8h;
typedef __attribute__((ext_vector_type(8)))  float    v8f;

constexpr int NN  = 50000;   // nodes
constexpr int EE  = 800000;  // edges
constexpr int KIN = 128;     // input dim of every GEMM (IN == H*HID == 128)
constexpr int HD0 = 128;     // H*HID for layers 0..2
constexpr int CC  = 16;      // classes
constexpr int GG  = 32;      // graphs
constexpr float NEG_SLOPE = 0.2f;

// ---------------- float atomic max (init to -inf) ----------------
__device__ __forceinline__ void atomicMaxF(float* addr, float v) {
  if (v >= 0.0f) atomicMax((int*)addr, __float_as_int(v));
  else           atomicMin((unsigned int*)addr, __float_as_uint(v));
}

// ---------------- fill ----------------
__global__ void fill_f32(float* __restrict__ p, float v, int n) {
  int i = blockIdx.x * blockDim.x + threadIdx.x;
  if (i < n) p[i] = v;
}

// ---------------- f32 -> f16 convert (optionally fused relu) ----------------
__global__ void cvt_x_f16(const float* __restrict__ X, _Float16* __restrict__ Xh,
                          int n, int do_relu) {
  int i = blockIdx.x * blockDim.x + threadIdx.x;
  if (i >= n) return;
  float v = X[i];
  if (do_relu) v = fmaxf(v, 0.f);
  Xh[i] = (_Float16)v;
}

// ---------------- W[Kin,Kout] f32 -> Wt[Kout,Kin] f16 (transpose) ----------------
__global__ void cvt_w_t(const float* __restrict__ W, _Float16* __restrict__ Wt, int Kout) {
  int idx = blockIdx.x * blockDim.x + threadIdx.x;
  if (idx >= KIN * Kout) return;
  int k = idx / Kout, c = idx - k * Kout;
  Wt[(size_t)c * KIN + k] = (_Float16)W[idx];
}

// ---------------- WMMA GEMM: Z[N,Kout] = X[N,128] @ W[128,Kout] (f16 in, f32 out) ----
// X: [N,128] f16 row-major; Wt: [Kout,128] f16 (pre-transposed so B-columns are rows).
// 128 threads = 4 waves per block; each wave owns one 16x16 output tile.
// A fragment (16x32 f16, ISA 7.12.2): lane m=lane&15, grp=lane>>4;
//   elements 0..7 -> K = k0+grp*8+{0..7};  elements 8..15 -> K = k0+16+grp*8+{0..7}
// B fragment (32x16 f16): lane col n=lane&15; same K mapping as A
// Both are contiguous 16-byte chunks -> global_load_b128.
// C/D f32: VGPR r -> row = r + grp*8, col = lane&15
__global__ void gemm_wmma_f16(const _Float16* __restrict__ X, const _Float16* __restrict__ Wt,
                              float* __restrict__ Z, int Kout)
{
  const int lane = threadIdx.x & 31;
  const int wave = threadIdx.x >> 5;
  const int tm = blockIdx.x;            // 16-row tile
  const int tn = blockIdx.y * 4 + wave; // 16-col tile
  if (tn * 16 >= Kout) return;          // wave-uniform predicate
  const int m   = lane & 15;
  const int grp = lane >> 4;
  const int ncol = tn * 16 + m;
  const _Float16* xrow = X + (size_t)(tm * 16 + m) * KIN;
  const _Float16* wrow = Wt + (size_t)ncol * KIN;
  v8f acc = {};
#pragma unroll
  for (int k0 = 0; k0 < KIN; k0 += 32) {
    v8h a0 = *(const v8h*)(xrow + k0 + grp * 8);
    v8h a1 = *(const v8h*)(xrow + k0 + 16 + grp * 8);
    v8h b0 = *(const v8h*)(wrow + k0 + grp * 8);
    v8h b1 = *(const v8h*)(wrow + k0 + 16 + grp * 8);
    v16h a, b;
#pragma unroll
    for (int i = 0; i < 8; ++i) {
      a[i] = a0[i]; a[i + 8] = a1[i];
      b[i] = b0[i]; b[i + 8] = b1[i];
    }
    acc = __builtin_amdgcn_wmma_f32_16x16x32_f16(false, a, false, b,
                                                 (short)0, acc, false, false);
  }
#pragma unroll
  for (int r = 0; r < 8; ++r) {
    int row = tm * 16 + r + grp * 8;
    Z[(size_t)row * Kout + ncol] = acc[r];
  }
}

// ---------------- edge scoring: score[e,h] = attn_h . leakyrelu(z[src]+z[dst]) ----------------
// one wave (32 lanes) per edge; lanes stride the hd dims; per-head partials reduced via shfl_xor.
__global__ void edge_score(const float* __restrict__ z, const int* __restrict__ src,
                           const int* __restrict__ dst, const float* __restrict__ attn,
                           float* __restrict__ score, float* __restrict__ segm,
                           int nE, int hd, int d)
{
  int w = (blockIdx.x * blockDim.x + threadIdx.x) >> 5;
  int lane = threadIdx.x & 31;
  if (w >= nE) return;
  int s = src[w], t = dst[w];
  const float* zs = z + (size_t)s * hd;
  const float* zt = z + (size_t)t * hd;
  __builtin_prefetch(zs, 0, 0);
  __builtin_prefetch(zt, 0, 0);
  float s0 = 0.f, s1 = 0.f;
  for (int i = lane; i < hd; i += 32) {
    float v = zs[i] + zt[i];
    v = v > 0.f ? v : NEG_SLOPE * v;
    float p = v * attn[i];
    if (i < d) s0 += p; else s1 += p;
  }
#pragma unroll
  for (int off = 16; off > 0; off >>= 1) {
    s0 += __shfl_xor(s0, off, 32);
    s1 += __shfl_xor(s1, off, 32);
  }
  if (lane == 0) {
    score[2 * w]     = s0;
    score[2 * w + 1] = s1;
    atomicMaxF(&segm[2 * t],     s0);
    atomicMaxF(&segm[2 * t + 1], s1);
  }
}

// ---------------- a = exp(score - m[dst]); denom[dst] += a ----------------
__global__ void edge_exp(const int* __restrict__ dst, float* __restrict__ score,
                         const float* __restrict__ segm, float* __restrict__ segd, int nE)
{
  int idx = blockIdx.x * blockDim.x + threadIdx.x;
  if (idx >= 2 * nE) return;
  int e = idx >> 1, h = idx & 1;
  int t = dst[e];
  float a = expf(score[idx] - segm[2 * t + h]);
  score[idx] = a;
  atomicAdd(&segd[2 * t + h], a);
}

// ---------------- out[dst] += (a/denom[dst]) * z[src] ----------------
__global__ void edge_aggr(const float* __restrict__ z, const int* __restrict__ src,
                          const int* __restrict__ dst, const float* __restrict__ a,
                          const float* __restrict__ denom, float* __restrict__ outb,
                          int nE, int hd, int d)
{
  int w = (blockIdx.x * blockDim.x + threadIdx.x) >> 5;
  int lane = threadIdx.x & 31;
  if (w >= nE) return;
  int s = src[w], t = dst[w];
  float d0 = denom[2 * t];     d0 = (d0 == 0.f) ? 1.f : d0;
  float d1 = denom[2 * t + 1]; d1 = (d1 == 0.f) ? 1.f : d1;
  float al0 = a[2 * w] / d0;
  float al1 = a[2 * w + 1] / d1;
  const float* zs = z + (size_t)s * hd;
  float* ot = outb + (size_t)t * hd;
  for (int i = lane; i < hd; i += 32) {
    float al = (i < d) ? al0 : al1;
    atomicAdd(&ot[i], al * zs[i]);
  }
}

// ---------------- mean over 2 heads: [N,2,C] -> [N,C] ----------------
__global__ void mean_heads_k(const float* __restrict__ ob, float* __restrict__ lg, int n, int c) {
  int idx = blockIdx.x * blockDim.x + threadIdx.x;
  if (idx >= n * c) return;
  int r = idx / c, j = idx - r * c;
  lg[idx] = 0.5f * (ob[(size_t)r * 2 * c + j] + ob[(size_t)r * 2 * c + c + j]);
}

// ---------------- graph pooling: pooled[g] += logits[n] ----------------
__global__ void pool_k(const float* __restrict__ lg, const int* __restrict__ ng,
                       float* __restrict__ pooled, int n, int c) {
  int idx = blockIdx.x * blockDim.x + threadIdx.x;
  if (idx >= n * c) return;
  int r = idx / c, j = idx - r * c;
  atomicAdd(&pooled[ng[r] * c + j], lg[idx]);
}

// ---------------- MLP head: lin -> batchnorm(train) -> relu -> lin -> log_softmax ----------------
// one block, 512 threads (g = t>>4 in [0,32), c = t&15 in [0,16))
__global__ void mlp_k(const float* __restrict__ pooled, const float* __restrict__ w1,
                      const float* __restrict__ gamma, const float* __restrict__ beta,
                      const float* __restrict__ w2, float* __restrict__ out)
{
  __shared__ float h1[GG][CC];
  __shared__ float mu[CC], iv[CC];
  __shared__ float h2[GG][CC];
  int t = threadIdx.x;
  int g = t >> 4, c = t & 15;
  float acc = 0.f;
  for (int k = 0; k < CC; ++k) acc += pooled[g * CC + k] * w1[k * CC + c];
  h1[g][c] = acc;
  __syncthreads();
  if (t < CC) {
    float m = 0.f;
    for (int i = 0; i < GG; ++i) m += h1[i][t];
    m /= (float)GG;
    float v = 0.f;
    for (int i = 0; i < GG; ++i) { float dd = h1[i][t] - m; v += dd * dd; }
    v /= (float)GG;
    mu[t] = m;
    iv[t] = 1.0f / sqrtf(v + 1e-5f);
  }
  __syncthreads();
  float hn = (h1[g][c] - mu[c]) * iv[c] * gamma[c] + beta[c];
  __syncthreads();
  h1[g][c] = hn > 0.f ? hn : 0.f;
  __syncthreads();
  acc = 0.f;
  for (int k = 0; k < CC; ++k) acc += h1[g][k] * w2[k * CC + c];
  h2[g][c] = acc;
  __syncthreads();
  if (t < GG) {
    float mx = -INFINITY;
    for (int k = 0; k < CC; ++k) mx = fmaxf(mx, h2[t][k]);
    float s = 0.f;
    for (int k = 0; k < CC; ++k) s += expf(h2[t][k] - mx);
    float lse = logf(s) + mx;
    for (int k = 0; k < CC; ++k) out[t * CC + k] = h2[t][k] - lse;
  }
}

// ---------------- one GATv2 layer ----------------
// relu_in: apply relu of the *previous* layer while converting Xin to f16
static void run_gat_layer(const float* Xin, const float* Wl, const float* attnl,
                          int hd, int d, bool relu_in,
                          const int* src, const int* dst,
                          _Float16* xh, _Float16* wh,
                          float* zbuf, float* outb, float* score,
                          float* segm, float* segd, hipStream_t s)
{
  cvt_x_f16<<<(NN * KIN + 255) / 256, 256, 0, s>>>(Xin, xh, NN * KIN, relu_in ? 1 : 0);
  cvt_w_t<<<(KIN * hd + 255) / 256, 256, 0, s>>>(Wl, wh, hd);
  gemm_wmma_f16<<<dim3((unsigned)(NN / 16), (unsigned)((hd + 63) / 64)), 128, 0, s>>>(
      xh, wh, zbuf, hd);
  fill_f32<<<(2 * NN + 255) / 256, 256, 0, s>>>(segm, -INFINITY, 2 * NN);
  fill_f32<<<(2 * NN + 255) / 256, 256, 0, s>>>(segd, 0.f, 2 * NN);
  fill_f32<<<(NN * hd + 255) / 256, 256, 0, s>>>(outb, 0.f, NN * hd);
  edge_score<<<(EE + 7) / 8, 256, 0, s>>>(zbuf, src, dst, attnl, score, segm, EE, hd, d);
  edge_exp<<<(2 * EE + 255) / 256, 256, 0, s>>>(dst, score, segm, segd, EE);
  edge_aggr<<<(EE + 7) / 8, 256, 0, s>>>(zbuf, src, dst, score, segd, outb, EE, hd, d);
}

extern "C" void kernel_launch(void* const* d_in, const int* in_sizes, int n_in,
                              void* d_out, int out_size, void* d_ws, size_t ws_size,
                              hipStream_t stream) {
  const float* feat  = (const float*)d_in[0];
  const float* W0    = (const float*)d_in[1];
  const float* attn0 = (const float*)d_in[2];
  const float* W1    = (const float*)d_in[3];
  const float* attn1 = (const float*)d_in[4];
  const float* W2    = (const float*)d_in[5];
  const float* attn2 = (const float*)d_in[6];
  const float* Wf    = (const float*)d_in[7];
  const float* attnf = (const float*)d_in[8];
  const float* mw1   = (const float*)d_in[9];
  const float* mg    = (const float*)d_in[10];
  const float* mb    = (const float*)d_in[11];
  const float* mw2   = (const float*)d_in[12];
  const int*   src   = (const int*)d_in[13];
  const int*   dst   = (const int*)d_in[14];
  const int*   ngr   = (const int*)d_in[15];
  float* out = (float*)d_out;

  float* ws      = (float*)d_ws;
  float* zbuf    = ws;                            // N*128 f32
  float* hbuf    = zbuf + (size_t)NN * HD0;       // N*128 f32 (activations, reused in place)
  float* score   = hbuf + (size_t)NN * HD0;       // E*2
  float* segm    = score + (size_t)EE * 2;        // N*2
  float* segd    = segm + (size_t)NN * 2;         // N*2
  float* logitsb = segd + (size_t)NN * 2;         // N*16
  float* pooled  = logitsb + (size_t)NN * CC;     // G*16
  _Float16* xh   = (_Float16*)(pooled + GG * CC); // N*128 f16 (16B-aligned offset)
  _Float16* wh   = xh + (size_t)NN * KIN;         // 128*128 f16

  // 3 hidden GATv2 layers (H=2, D=64) + final GATv2 layer (H=2, D=16)
  run_gat_layer(feat, W0, attn0, HD0, 64, false, src, dst, xh, wh,
                zbuf, hbuf, score, segm, segd, stream);
  run_gat_layer(hbuf, W1, attn1, HD0, 64, true,  src, dst, xh, wh,
                zbuf, hbuf, score, segm, segd, stream);
  run_gat_layer(hbuf, W2, attn2, HD0, 64, true,  src, dst, xh, wh,
                zbuf, hbuf, score, segm, segd, stream);
  run_gat_layer(hbuf, Wf, attnf, 2 * CC, CC, true, src, dst, xh, wh,
                zbuf, hbuf, score, segm, segd, stream);

  mean_heads_k<<<(NN * CC + 255) / 256, 256, 0, stream>>>(hbuf, logitsb, NN, CC);
  fill_f32<<<(GG * CC + 255) / 256, 256, 0, stream>>>(pooled, 0.f, GG * CC);
  pool_k<<<(NN * CC + 255) / 256, 256, 0, stream>>>(logitsb, ngr, pooled, NN, CC);
  mlp_k<<<1, GG * CC, 0, stream>>>(pooled, mw1, mg, mb, mw2, out);
}
